// BoundaryPathFinder_5394478924371
// MI455X (gfx1250) — matrix operand order repaired and checked by reading, not compiled, not run on hardware
//
#include <hip/hip_runtime.h>
#include <stdint.h>

typedef _Float16 v16h __attribute__((ext_vector_type(16)));
typedef float    v8f  __attribute__((ext_vector_type(8)));

constexpr int IMG  = 224;
constexpr int PIX  = IMG * IMG;       // 50176
constexpr int NB   = 256;             // batch
constexpr int PPD  = 7;               // paths per direction
constexpr int NP   = 11;              // band width 2*5+1
constexpr int BW_  = 5;
constexpr int KSEG = 64;
constexpr int WAVES_PER_IMG = 16;
constexpr int CHUNKS = PIX / 32;                       // 1568
constexpr int CHUNKS_PER_WAVE = CHUNKS / WAVES_PER_IMG; // 98

// ---------------------------------------------------------------------------
// Kernel 1: banded seam-carving DP. One wave per (image, direction, path).
// Lanes 0..10 hold the 11 band positions; sequential over 224 rows with
// neighbor costs exchanged via wave32 shuffles. Backpointers in LDS, lane 0
// backtracks and writes the path (column index per row) as u16.
// ---------------------------------------------------------------------------
__global__ __launch_bounds__(32) void seam_kernel(const float* __restrict__ g,
                                                  unsigned short* __restrict__ paths) {
  const int gb  = blockIdx.x;           // b*14 + dir*7 + p
  const int b   = gb / 14;
  const int rem = gb % 14;
  const int dir = rem / PPD;            // 0 = vertical (rows as steps), 1 = horizontal
  const int p   = rem % PPD;
  const float* gi = g + (size_t)b * PIX;
  const int init = (p + 1) * (IMG / 8); // 28, 56, ..., 196
  const int j    = threadIdx.x;
  const bool act = (j < NP);
  const int col  = min(max(init - BW_ + j, 0), IMG - 1);

  __shared__ unsigned char bp[IMG][12];
  __shared__ float fcost[NP];

  float cost = 0.0f;
  if (act) cost = -((dir == 0) ? gi[col] : gi[col * IMG]);

  const int jm = (j - 1 < 0) ? 0 : ((j - 1 > NP - 1) ? (NP - 1) : (j - 1));
  const int jp = (j + 1 > NP - 1) ? (NP - 1) : (j + 1);

  for (int s = 1; s < IMG; ++s) {
    float c = 0.0f;
    if (act) c = -((dir == 0) ? gi[s * IMG + col] : gi[col * IMG + s]);
    const float cm = __shfl(cost, jm, 32);
    const float cp = __shfl(cost, jp, 32);
    float mn; int bpj;
    // first-occurrence argmin over [cost[jm], cost[j], cost[jp]] (matches jnp.argmin)
    if (cm <= cost && cm <= cp) { mn = cm;   bpj = jm; }
    else if (cost <= cp)        { mn = cost; bpj = j;  }
    else                        { mn = cp;   bpj = jp; }
    cost = mn + c;
    if (act) bp[s][j] = (unsigned char)bpj;
  }
  if (act) fcost[j] = cost;
  __syncthreads();
  if (j == 0) {
    int idx = 0; float best = fcost[0];
    for (int q = 1; q < NP; ++q)
      if (fcost[q] < best) { best = fcost[q]; idx = q; }
    unsigned short* out = paths + (size_t)gb * IMG;
    for (int s = IMG - 1; s >= 1; --s) {
      out[s] = (unsigned short)min(max(init - BW_ + idx, 0), IMG - 1);
      idx = bp[s][idx];
    }
    out[0] = (unsigned short)min(max(init - BW_ + idx, 0), IMG - 1);
  }
}

// ---------------------------------------------------------------------------
// Kernel 2: labels. One block per (image, row); thread x = column.
// vlab(y,x) = #distinct vertical-path columns <= x on this row;
// hlab(y,x) = #distinct horizontal-path rows <= y on this column.
// label = vlab + 8*hlab. Writes u8 labels (scratch) + float mask (output).
// ---------------------------------------------------------------------------
__global__ __launch_bounds__(IMG) void label_kernel(const unsigned short* __restrict__ paths,
                                                    unsigned char* __restrict__ labels,
                                                    float* __restrict__ mask_out) {
  const int b = blockIdx.x / IMG;
  const int y = blockIdx.x % IMG;
  const int x = threadIdx.x;
  const unsigned short* vp = paths + (size_t)(b * 14)     * IMG; // [p][row]
  const unsigned short* hp = paths + (size_t)(b * 14 + 7) * IMG; // [p][col]
  int v[PPD], h[PPD];
#pragma unroll
  for (int p = 0; p < PPD; ++p) { v[p] = vp[p * IMG + y]; h[p] = hp[p * IMG + x]; }
  int vlab = 0, hlab = 0;
#pragma unroll
  for (int p = 0; p < PPD; ++p) {
    bool dv = false, dh = false;
#pragma unroll
    for (int q = 0; q < PPD; ++q) {
      if (q < p) { dv = dv || (v[q] == v[p]); dh = dh || (h[q] == h[p]); }
    }
    if (!dv && v[p] <= x) ++vlab;
    if (!dh && h[p] <= y) ++hlab;
  }
  const int lab = vlab + 8 * hlab;
  const size_t off = (size_t)b * PIX + (size_t)y * IMG + x;
  labels[off]   = (unsigned char)lab;
  mask_out[off] = (float)lab;
}

__global__ void zero_kernel(float* __restrict__ p, int n) {
  const int i = blockIdx.x * blockDim.x + threadIdx.x;
  if (i < n) p[i] = 0.0f;
}

// ---------------------------------------------------------------------------
// Kernel 3: segment sums as one-hot matmul on WMMA.
// Per 32-pixel chunk: A (16x32 f16) = one-hot(label == m+16t) built per the
// contiguous k order k = 16*grp + h; B (32x16 f16) cols {0,1,2} = (y, x, 1) per the
// same k order; C (16x16 f32) accumulates (sy, sx, cnt) per label.
// 4 tiles cover the 64 labels. Exact in f16/f32 (coords <= 223, sums < 2^24).
// ---------------------------------------------------------------------------
__global__ __launch_bounds__(32) void centroid_wmma_kernel(const unsigned char* __restrict__ labels,
                                                           float* __restrict__ partials) {
  const int b    = blockIdx.x / WAVES_PER_IMG;
  const int w    = blockIdx.x % WAVES_PER_IMG;
  const int lane = threadIdx.x;
  const unsigned char* lb = labels + (size_t)b * PIX;

  const int n   = lane & 15;          // C column / B column
  const int m   = lane & 15;          // A row
  const int grp = (lane >> 4) & 1;    // half-wave group

  v8f acc[4] = {v8f{}, v8f{}, v8f{}, v8f{}};

  for (int c = 0; c < CHUNKS_PER_WAVE; ++c) {
    const int p0 = (w * CHUNKS_PER_WAVE + c) * 32;
    const int myLab = (int)lb[p0 + lane];   // coalesced: lane k holds label of pixel p0+k

    // Gather the 16 labels this lane's A-fragment needs (A 16x32 layout):
    // K = grp*16 + h
    int lv[16];
#pragma unroll
    for (int h = 0; h < 16; ++h) {
      const int K = h + (grp ? 16 : 0);
      lv[h] = __shfl(myLab, K, 32);
    }

    // B fragment: rows K = grp*16 + h, column n; values (y, x, 1, 0...)
    v16h bf;
#pragma unroll
    for (int h = 0; h < 16; ++h) {
      const unsigned pix = (unsigned)(p0 + (grp ? 16 : 0) + h);
      const unsigned yy  = pix / (unsigned)IMG;
      const unsigned xx  = pix - yy * (unsigned)IMG;
      const float bv = (n == 0) ? (float)yy : (n == 1) ? (float)xx
                     : (n == 2) ? 1.0f : 0.0f;
      bf[h] = (_Float16)bv;
    }

#pragma unroll
    for (int t = 0; t < 4; ++t) {
      v16h af;
#pragma unroll
      for (int h = 0; h < 16; ++h)
        af[h] = (lv[h] == (m + 16 * t)) ? (_Float16)1.0f : (_Float16)0.0f;
      acc[t] = __builtin_amdgcn_wmma_f32_16x16x32_f16(
          /*neg_a=*/false, af, /*neg_b=*/false, bf,
          /*c_mod=*/(short)0, acc[t], /*reuse_a=*/false, /*reuse_b=*/false);
    }
  }

  // C layout: VGPR r, lanes 0-15 -> M=r (N=lane), lanes 16-31 -> M=8+r (N=lane-16)
  float* part = partials + (size_t)b * (KSEG * 3);
  if (n < 3) {
#pragma unroll
    for (int t = 0; t < 4; ++t) {
#pragma unroll
      for (int r = 0; r < 8; ++r) {
        const int k = 16 * t + (grp ? (8 + r) : r);
        atomicAdd(&part[k * 3 + n], acc[t][r]);
      }
    }
  }
}

// ---------------------------------------------------------------------------
// Kernel 4: centroids = sums / max(cnt, 1e-6)
// ---------------------------------------------------------------------------
__global__ void cent_kernel(const float* __restrict__ partials, float* __restrict__ cents) {
  const int i = blockIdx.x * blockDim.x + threadIdx.x;
  if (i >= NB * KSEG) return;
  const int b = i / KSEG, k = i % KSEG;
  const float* p = partials + (size_t)b * (KSEG * 3);
  const float sy  = p[k * 3 + 0];
  const float sx  = p[k * 3 + 1];
  const float cnt = fmaxf(p[k * 3 + 2], 1e-6f);
  cents[(size_t)b * KSEG * 2 + k * 2 + 0] = sy / cnt;
  cents[(size_t)b * KSEG * 2 + k * 2 + 1] = sx / cnt;
}

extern "C" void kernel_launch(void* const* d_in, const int* in_sizes, int n_in,
                              void* d_out, int out_size, void* d_ws, size_t ws_size,
                              hipStream_t stream) {
  (void)in_sizes; (void)n_in; (void)out_size; (void)ws_size;
  // inputs: d_in[0] = x (unused by reference), d_in[1] = grad_map (B,224,224) f32
  const float* grad = (const float*)d_in[1];

  float* out      = (float*)d_out;
  float* cents    = out;                         // B*64*2 floats
  float* mask_out = out + NB * KSEG * 2;         // B*224*224 floats

  // workspace layout
  unsigned char* wsb = (unsigned char*)d_ws;
  unsigned short* paths = (unsigned short*)wsb;                  // B*14*224 u16
  const size_t paths_bytes  = (size_t)NB * 14 * IMG * sizeof(unsigned short);
  unsigned char* labels = wsb + paths_bytes;                     // B*PIX u8
  const size_t labels_bytes = (size_t)NB * PIX;
  float* partials = (float*)(wsb + paths_bytes + labels_bytes);  // B*64*3 f32

  seam_kernel<<<NB * 14, 32, 0, stream>>>(grad, paths);
  label_kernel<<<NB * IMG, IMG, 0, stream>>>(paths, labels, mask_out);
  const int npart = NB * KSEG * 3;
  zero_kernel<<<(npart + 255) / 256, 256, 0, stream>>>(partials, npart);
  centroid_wmma_kernel<<<NB * WAVES_PER_IMG, 32, 0, stream>>>(labels, partials);
  cent_kernel<<<(NB * KSEG + 255) / 256, 256, 0, stream>>>(partials, cents);
}
